// EMAQuantizer_21474836480186
// MI455X (gfx1250) — matrix operand order repaired and checked by reading, not compiled, not run on hardware
//
#include <hip/hip_runtime.h>

typedef __attribute__((ext_vector_type(16))) __bf16 v16bf;
typedef __attribute__((ext_vector_type(8)))  __bf16 v8bf;
typedef __attribute__((ext_vector_type(8)))  float  v8f;

#define N_IN   131072
#define DIM    256
#define KCODES 1024

// Async global->LDS copy (16B per lane). Uses inline asm: the clang builtin's
// pointer params are typed in the CUDA __device__/__shared__ language address
// spaces which HIP source cannot spell, so we bypass it. vDst = LDS byte
// address (low 32 bits of the generic shared pointer), vAddr = 64-bit global
// address, saddr = off (GV mode). Tracked on ASYNCcnt.
__device__ __forceinline__ void async_cp16(const void* g, void* l) {
  asm volatile("global_load_async_to_lds_b128 %0, %1, off"
               :: "v"((unsigned)(size_t)l),
                  "v"((unsigned long long)(size_t)g)
               : "memory");
}

__device__ __forceinline__ void wait_async_zero() {
#if defined(__AMDGCN__) && __has_builtin(__builtin_amdgcn_s_wait_asynccnt)
  __builtin_amdgcn_s_wait_asynccnt(0);
#else
  asm volatile("s_wait_asynccnt 0x0" ::: "memory");
#endif
}

// ---------------- output layout (floats) ----------------
// 0        : quantized_st  [N*D]          = 33554432
// 33554432 : loss          [1]
// 33554433 : indices       [N]            = 131072
// 33685505 : new_embedding [K*D]          = 262144
// 33947649 : new_cluster   [K]            = 1024
// 33948673 : new_emb_avg   [K*D]          = 262144
// total 34210817

// ---------------- workspace layout (bytes) ----------------
// 0        : x_bf   [N*D] bf16 = 67108864
// 67108864 : e_bf   [K*D] bf16 = 524288
// 67633152 : xnorm  [N]   f32  = 524288
// 68157440 : enorm  [K]   f32  = 4096
// 68161536 : idx    [N]   i32  = 524288
// 68685824 : counts [K]   f32  = 4096      \
// 68689920 : embsum [K*D] f32  = 1048576    > contiguous, zeroed per launch
// 69738496 : scal   [4]   f32  (sqsum, ortho, ntotal)  /

__global__ __launch_bounds__(256) void k_zero(float* __restrict__ p, int n) {
  int i = blockIdx.x * 256 + threadIdx.x;
  if (i < n) p[i] = 0.0f;
}

// one wave per row: f32 -> bf16 convert + squared-norm reduction
__global__ __launch_bounds__(256) void k_prep(const float* __restrict__ src,
                                              __bf16* __restrict__ dst,
                                              float* __restrict__ norms) {
  const int lane = threadIdx.x & 31;
  const int wave = threadIdx.x >> 5;
  const int row  = blockIdx.x * 8 + wave;
  const float4* p = reinterpret_cast<const float4*>(src + (size_t)row * DIM + lane * 8);
  float4 v0 = p[0], v1 = p[1];
  float ss = v0.x*v0.x + v0.y*v0.y + v0.z*v0.z + v0.w*v0.w
           + v1.x*v1.x + v1.y*v1.y + v1.z*v1.z + v1.w*v1.w;
  v8bf h;
  h[0] = (__bf16)v0.x; h[1] = (__bf16)v0.y; h[2] = (__bf16)v0.z; h[3] = (__bf16)v0.w;
  h[4] = (__bf16)v1.x; h[5] = (__bf16)v1.y; h[6] = (__bf16)v1.z; h[7] = (__bf16)v1.w;
  *reinterpret_cast<v8bf*>(dst + (size_t)row * DIM + lane * 8) = h;
#pragma unroll
  for (int off = 16; off > 0; off >>= 1) ss += __shfl_xor(ss, off, 32);
  if (lane == 0) norms[row] = ss;
}

// WMMA distance + argmin. Block = 8 waves; wave owns 16 rows x all 1024 codes.
// A (16x32 bf16) VGPR layout (contiguous K): lane<16 -> row M=lane, K = kb+{0..15};
// lane>=16 -> row M=lane-16, K = kb+{16..31}. B tiles (16 codes x 256 k,
// 8 KB) are staged in LDS once per block (shared by all 8 waves; cuts L2 traffic
// 8x), double-buffered via async global->LDS copies. B fragment: lane holds
// column n = nb*16+(lane&15) with 16 consecutive K -> one 32B ds read pair;
// row stride padded to 528B so the 16 lanes hit a clean bank permutation.
__global__ __launch_bounds__(256) void k_dist(const __bf16* __restrict__ xbf,
                                              const __bf16* __restrict__ ebf,
                                              const float* __restrict__ xnorm,
                                              const float* __restrict__ enorm,
                                              int* __restrict__ idx) {
  constexpr int BSTRIDE = 264;                 // 256 + 8 bf16 pad (16B)
  __shared__ __bf16 smem[2][16 * BSTRIDE];

  const int lane = threadIdx.x & 31;
  const int wave = threadIdx.x >> 5;
  const int half = lane >> 4;
  const int l15  = lane & 15;
  const int mr   = blockIdx.x * 128 + wave * 16;
  const int m    = mr + l15;

  // cooperative copy mapping: thread t moves 32B of the 8KB tile
  const int cc = threadIdx.x >> 4;             // code-in-tile 0..15
  const int ck = (threadIdx.x & 15) * 16;      // element offset 0..240

  auto stage = [&](int nb, int buf) {
    const __bf16* g = ebf + (size_t)(nb * 16 + cc) * DIM + ck;
    __bf16* l = &smem[buf][cc * BSTRIDE + ck];
    async_cp16(g, l);
    async_cp16(g + 8, l + 8);
  };

  // A slab: 16 rows x 256 k held in registers for the whole kernel (64 VGPRs)
  v16bf a[8];
#pragma unroll
  for (int kk = 0; kk < 8; ++kk) {
    const int kb = kk * 32;
    union { v16bf v; v8bf h[2]; } u;
    u.h[0] = *reinterpret_cast<const v8bf*>(xbf + (size_t)m * DIM + kb + half * 16);
    u.h[1] = *reinterpret_cast<const v8bf*>(xbf + (size_t)m * DIM + kb + half * 16 + 8);
    a[kk] = u.v;
  }

  float xn[8];
#pragma unroll
  for (int j = 0; j < 8; ++j) xn[j] = xnorm[mr + half * 8 + j];

  float bestd[8];
  int   besti[8];
#pragma unroll
  for (int j = 0; j < 8; ++j) { bestd[j] = 3.4e38f; besti[j] = 0; }

  stage(0, 0);
  for (int nb = 0; nb < KCODES / 16; ++nb) {
    const int buf = nb & 1;
    wait_async_zero();                          // tile nb landed in LDS
    __syncthreads();                            // visible to all waves; prior
                                                // reads of buf^1 are done
    if (nb + 1 < KCODES / 16) stage(nb + 1, buf ^ 1);

    const int n = nb * 16 + l15;
    const float en = enorm[n];
    v8f acc = {};
    const __bf16* brow = &smem[buf][l15 * BSTRIDE + half * 16];
#pragma unroll
    for (int kk = 0; kk < 8; ++kk) {
      union { v16bf v; v8bf h[2]; } ub;
      const __bf16* bp = brow + kk * 32;
      ub.h[0] = *reinterpret_cast<const v8bf*>(bp);
      ub.h[1] = *reinterpret_cast<const v8bf*>(bp + 8);
      acc = __builtin_amdgcn_wmma_f32_16x16x32_bf16(false, a[kk], false, ub.v,
                                                    (short)0, acc, false, false);
    }
#pragma unroll
    for (int j = 0; j < 8; ++j) {
      float dist = xn[j] + en - 2.0f * acc[j];
      if (dist < bestd[j]) { bestd[j] = dist; besti[j] = n; }
    }
  }

  // argmin across the 16 lanes holding the same row (lower index wins ties)
#pragma unroll
  for (int j = 0; j < 8; ++j) {
#pragma unroll
    for (int off = 1; off < 16; off <<= 1) {
      float od = __shfl_xor(bestd[j], off, 16);
      int   oi = __shfl_xor(besti[j], off, 16);
      if (od < bestd[j] || (od == bestd[j] && oi < besti[j])) {
        bestd[j] = od; besti[j] = oi;
      }
    }
  }
  if (l15 == 0) {
#pragma unroll
    for (int j = 0; j < 8; ++j) idx[mr + half * 8 + j] = besti[j];
  }
}

// gather codebook row, straight-through output, segment sums, (q-x)^2 reduction
__global__ __launch_bounds__(256) void k_quant(const float* __restrict__ x,
                                               const float* __restrict__ emb,
                                               const int* __restrict__ idx,
                                               float* __restrict__ out_q,
                                               float* __restrict__ out_idx,
                                               float* __restrict__ counts,
                                               float* __restrict__ emb_sum,
                                               float* __restrict__ sq_acc) {
  __shared__ float red[256];
  const size_t gid = (size_t)blockIdx.x * 256 + threadIdx.x;
  const int row = (int)(gid >> 8);
  const int d   = (int)(gid & 255);
  const int k   = idx[row];
  const float xv = x[gid];
  const float qv = emb[(size_t)k * DIM + d];
  const float dq = qv - xv;
  out_q[gid] = xv + dq;
  if (d == 0) {
    out_idx[row] = (float)k;
    atomicAdd(&counts[k], 1.0f);
  }
  atomicAdd(&emb_sum[(size_t)k * DIM + d], xv);
  red[threadIdx.x] = dq * dq;
  __syncthreads();
  for (int s = 128; s > 0; s >>= 1) {
    if (threadIdx.x < s) red[threadIdx.x] += red[threadIdx.x + s];
    __syncthreads();
  }
  if (threadIdx.x == 0) atomicAdd(sq_acc, red[0]);
}

// sum over i!=j of (e_i . e_j)^2  (full off-diagonal, both orders)
__global__ __launch_bounds__(256) void k_ortho(const float* __restrict__ emb,
                                               float* __restrict__ ortho_acc) {
  __shared__ float red[256];
  const int p = blockIdx.x * 256 + threadIdx.x;
  const int i = p >> 10;
  const int j = p & 1023;
  float v = 0.0f;
  if (i != j) {
    const float4* a = reinterpret_cast<const float4*>(emb + (size_t)i * DIM);
    const float4* b = reinterpret_cast<const float4*>(emb + (size_t)j * DIM);
    float dot = 0.0f;
#pragma unroll 4
    for (int t = 0; t < DIM / 4; ++t) {
      float4 av = a[t], bv = b[t];
      dot += av.x * bv.x + av.y * bv.y + av.z * bv.z + av.w * bv.w;
    }
    v = dot * dot;
  }
  red[threadIdx.x] = v;
  __syncthreads();
  for (int s = 128; s > 0; s >>= 1) {
    if (threadIdx.x < s) red[threadIdx.x] += red[threadIdx.x + s];
    __syncthreads();
  }
  if (threadIdx.x == 0) atomicAdd(ortho_acc, red[0]);
}

__global__ __launch_bounds__(256) void k_ema_avg(const float* __restrict__ ema_avg,
                                                 const float* __restrict__ emb_sum,
                                                 float* __restrict__ out_avg) {
  const size_t gid = (size_t)blockIdx.x * 256 + threadIdx.x;
  out_avg[gid] = 0.99f * ema_avg[gid] + 0.01f * emb_sum[gid];
}

__global__ __launch_bounds__(256) void k_ema_cs(const float* __restrict__ ema_cs,
                                                const float* __restrict__ counts,
                                                float* __restrict__ out_cs,
                                                float* __restrict__ ntotal_acc) {
  __shared__ float red[256];
  const int k = blockIdx.x * 256 + threadIdx.x;
  const float ncs = 0.99f * ema_cs[k] + 0.01f * counts[k];
  out_cs[k] = ncs;
  red[threadIdx.x] = ncs;
  __syncthreads();
  for (int s = 128; s > 0; s >>= 1) {
    if (threadIdx.x < s) red[threadIdx.x] += red[threadIdx.x + s];
    __syncthreads();
  }
  if (threadIdx.x == 0) atomicAdd(ntotal_acc, red[0]);
}

__global__ __launch_bounds__(256) void k_final(const float* __restrict__ out_avg,
                                               const float* __restrict__ out_cs,
                                               const float* __restrict__ scal,
                                               float* __restrict__ out_emb,
                                               float* __restrict__ out_loss) {
  const size_t gid = (size_t)blockIdx.x * 256 + threadIdx.x;
  const int k = (int)(gid >> 8);
  const float nt = scal[2] + 1e-5f;
  const float cs = (out_cs[k] + 1e-5f) / nt;
  out_emb[gid] = out_avg[gid] / cs;
  if (gid == 0) {
    // q_latent_loss and e_latent_loss are numerically identical => 1.25 * mean
    const float mse = scal[0] / (float)((size_t)N_IN * DIM);
    out_loss[0] = 1.25f * mse + 0.09f * sqrtf(scal[1]);
  }
}

extern "C" void kernel_launch(void* const* d_in, const int* in_sizes, int n_in,
                              void* d_out, int out_size, void* d_ws, size_t ws_size,
                              hipStream_t stream) {
  (void)in_sizes; (void)n_in; (void)out_size; (void)ws_size;
  const float* x       = (const float*)d_in[0];
  const float* emb     = (const float*)d_in[1];
  const float* ema_cs  = (const float*)d_in[2];
  const float* ema_avg = (const float*)d_in[3];

  float* out      = (float*)d_out;
  float* out_q    = out;
  float* out_loss = out + 33554432;
  float* out_idx  = out + 33554433;
  float* out_emb  = out + 33685505;
  float* out_cs   = out + 33947649;
  float* out_avg  = out + 33948673;

  char* ws = (char*)d_ws;
  __bf16* xbf   = (__bf16*)(ws);
  __bf16* ebf   = (__bf16*)(ws + 67108864);
  float* xnorm  = (float*)(ws + 67633152);
  float* enorm  = (float*)(ws + 68157440);
  int*   idx    = (int*)(ws + 68161536);
  float* counts = (float*)(ws + 68685824);
  float* embsum = (float*)(ws + 68689920);
  float* scal   = (float*)(ws + 69738496);

  // zero accumulators (counts + embsum + scalars are contiguous: 263172 floats)
  k_zero<<<1029, 256, 0, stream>>>(counts, 263172);
  k_prep<<<N_IN / 8, 256, 0, stream>>>(x, xbf, xnorm);
  k_prep<<<KCODES / 8, 256, 0, stream>>>(emb, ebf, enorm);
  k_dist<<<N_IN / 128, 256, 0, stream>>>(xbf, ebf, xnorm, enorm, idx);
  k_quant<<<(N_IN * DIM) / 256, 256, 0, stream>>>(x, emb, idx, out_q, out_idx,
                                                  counts, embsum, &scal[0]);
  k_ortho<<<(KCODES * KCODES) / 256, 256, 0, stream>>>(emb, &scal[1]);
  k_ema_avg<<<(KCODES * DIM) / 256, 256, 0, stream>>>(ema_avg, embsum, out_avg);
  k_ema_cs<<<KCODES / 256, 256, 0, stream>>>(ema_cs, counts, out_cs, &scal[2]);
  k_final<<<(KCODES * DIM) / 256, 256, 0, stream>>>(out_avg, out_cs, scal,
                                                    out_emb, out_loss);
}
